// GATLayer_68101001446097
// MI455X (gfx1250) — matrix operand order repaired and checked by reading, not compiled, not run on hardware
//
#include <hip/hip_runtime.h>
#include <hip/hip_bf16.h>
#include <math.h>

// ---------------------------------------------------------------------------
// GAT layer for MI455X (gfx1250, wave32, WMMA, async global->LDS).
// B=4, N=1024, T=12, F_IN=64, H=64.
// ---------------------------------------------------------------------------

#define B_  4
#define N_  1024
#define T_  12
#define F_  64
#define H_  64
#define BT_ (B_ * T_)          // 48
#define M_  (BT_ * N_)         // 49152 total (b,t,n) rows

typedef __attribute__((ext_vector_type(16))) _Float16 v16h;
typedef __attribute__((ext_vector_type(8)))  _Float16 v8h;
typedef __attribute__((ext_vector_type(8)))  float    v8f;

// Combine two contiguous 8-half chunks into a 16-half fragment.
__device__ __forceinline__ v16h cat8(v8h lo, v8h hi) {
    return __builtin_shufflevector(lo, hi, 0, 1, 2, 3, 4, 5, 6, 7,
                                            8, 9, 10, 11, 12, 13, 14, 15);
}

// Async copy 16B global -> LDS (CDNA5 GLOBAL_LOAD_ASYNC_TO_LDS_B128, ASYNCcnt).
__device__ __forceinline__ void async_copy_b128(unsigned lds_off,
                                                unsigned long long gaddr) {
    asm volatile("global_load_async_to_lds_b128 %0, %1, off"
                 :: "v"(lds_off), "v"(gaddr) : "memory");
}

// ---------------------------------------------------------------------------
// Kernel 1: zero adjacency bitmask (32768 words)
// ---------------------------------------------------------------------------
__global__ void gat_zero_adj(unsigned int* __restrict__ adj) {
    adj[blockIdx.x * 256 + threadIdx.x] = 0u;
}

// ---------------------------------------------------------------------------
// Kernel 2: scatter edges into bitmask. edge_index is int64 [2, E].
// ---------------------------------------------------------------------------
__global__ void gat_edges(const long long* __restrict__ ei,
                          unsigned int* __restrict__ adj, int E) {
    int e = blockIdx.x * 256 + threadIdx.x;
    if (e >= E) return;
    int s = (int)ei[e];
    int d = (int)ei[E + e];
    atomicOr(adj + s * 32 + (d >> 5), 1u << (d & 31));
}

// ---------------------------------------------------------------------------
// Kernel 3: h = x_in @ W^T, output TRANSPOSED per (b,t): hbT[bt][hdim][node].
// Block = 128 (4 waves); block covers 64 rows (one bt); wave w owns rows
// [w*16, w*16+16) x all 64 output cols -> 4 n-tiles x 2 k-steps WMMA.
// A frag: two contiguous 8-half chunks per lane (ISA 7.12.2 16-bit A table).
// B frag: two contiguous 8-half chunks per lane (same K order as A) from N-major storage (W native layout).
// ---------------------------------------------------------------------------
__global__ void gat_gemm_h(const float* __restrict__ x,
                           const float* __restrict__ W,
                           _Float16* __restrict__ hbT) {
    __shared__ _Float16 Xs[64][F_];   // rows of x_in, f16          (8 KB)
    __shared__ _Float16 Wt[H_][F_];   // W as-is: Wt[h][f]          (8 KB)

    const int tid  = threadIdx.x;
    const int wave = tid >> 5;
    const int lane = tid & 31;
    const int half = lane >> 4;
    const int lm   = lane & 15;
    const int mbase    = blockIdx.x * 64;
    const int bt       = mbase / N_;       // whole block shares one (b,t)
    const int nodebase = mbase % N_;
    const int t = bt % T_;
    const int b = bt / T_;

    // Stage W (f32 -> f16), native [h][f] layout (N-major for the B operand).
    for (int i = tid; i < H_ * (F_ / 4); i += 128) {
        int hh = i >> 4, f4 = (i & 15) * 4;
        float4 v = *(const float4*)(W + hh * F_ + f4);
        Wt[hh][f4 + 0] = (_Float16)v.x;
        Wt[hh][f4 + 1] = (_Float16)v.y;
        Wt[hh][f4 + 2] = (_Float16)v.z;
        Wt[hh][f4 + 3] = (_Float16)v.w;
    }
    // Stage 64 x rows: x_in[b,t,n,f] = x[b,n,t,f]
    for (int i = tid; i < 64 * (F_ / 4); i += 128) {
        int r = i >> 4, f4 = (i & 15) * 4;
        int n = nodebase + r;
        float4 v = *(const float4*)(x + ((((size_t)b * N_ + n) * T_ + t) * F_ + f4));
        Xs[r][f4 + 0] = (_Float16)v.x;
        Xs[r][f4 + 1] = (_Float16)v.y;
        Xs[r][f4 + 2] = (_Float16)v.z;
        Xs[r][f4 + 3] = (_Float16)v.w;
    }
    __syncthreads();

    v8f acc[4] = {v8f{}, v8f{}, v8f{}, v8f{}};
#pragma unroll
    for (int ks = 0; ks < 2; ++ks) {
        v16h a = cat8(*(const v8h*)&Xs[wave * 16 + lm][ks * 32 + half * 8],
                      *(const v8h*)&Xs[wave * 16 + lm][ks * 32 + 16 + half * 8]);
#pragma unroll
        for (int nt = 0; nt < 4; ++nt) {
            v16h bf = cat8(*(const v8h*)&Wt[nt * 16 + lm][ks * 32 + half * 8],
                           *(const v8h*)&Wt[nt * 16 + lm][ks * 32 + 16 + half * 8]);
            acc[nt] = __builtin_amdgcn_wmma_f32_16x16x32_f16(
                false, a, false, bf, (short)0, acc[nt], false, false);
        }
    }

    // D layout: row m = v + 8*half (consecutive nodes over v) -> one b128/tile.
#pragma unroll
    for (int nt = 0; nt < 4; ++nt) {
        int hd    = nt * 16 + lm;
        int node0 = nodebase + wave * 16 + half * 8;
        v8h pk;
#pragma unroll
        for (int v = 0; v < 8; ++v) pk[v] = (_Float16)acc[nt][v];
        *(v8h*)(hbT + ((size_t)bt * H_ + hd) * N_ + node0) = pk;
    }
}

// ---------------------------------------------------------------------------
// Kernel 4: attn_l/attn_r dot products over hbT (coalesced across nodes).
// Grid (4, 48), block 256: one thread per node.
// ---------------------------------------------------------------------------
__global__ void gat_attn(const _Float16* __restrict__ hbT,
                         const float* __restrict__ a_l,
                         const float* __restrict__ a_r,
                         float* __restrict__ attn_l,
                         float* __restrict__ attn_r) {
    int bt   = blockIdx.y;
    int node = blockIdx.x * 256 + threadIdx.x;
    const _Float16* base = hbT + (size_t)bt * H_ * N_ + node;
    float sl = 0.f, sr = 0.f;
#pragma unroll
    for (int hd = 0; hd < H_; ++hd) {
        float hv = (float)base[(size_t)hd * N_];
        sl += hv * a_l[hd];
        sr += hv * a_r[hd];
    }
    attn_l[bt * N_ + node] = sl;
    attn_r[bt * N_ + node] = sr;
}

// ---------------------------------------------------------------------------
// Kernel 5: fused mask + leaky + softmax + WMMA aggregate + bias + GeLU.
// Grid: x = 16-row tile (64), y = bt (48). Block = 128 (4 waves).
// h tiles staged via double-buffered GLOBAL_LOAD_ASYNC_TO_LDS_B128.
// Softmax normalization deferred to the f32 epilogue (scale by rinv[row]).
// ---------------------------------------------------------------------------
__global__ void gat_agg(const _Float16* __restrict__ hbT,
                        const float* __restrict__ attn_l,
                        const float* __restrict__ attn_r,
                        const unsigned int* __restrict__ adj,
                        const float* __restrict__ bias,
                        float* __restrict__ out) {
    __shared__ _Float16 ebuf[16][N_];          // exp(e - max), f16   (32 KB)
    __shared__ float    ar_s[N_];              //                     ( 4 KB)
    __shared__ unsigned adj_s[16][32];         //                     ( 2 KB)
    __shared__ _Float16 hstage[2][H_][32];     // double-buffered B   ( 8 KB)
    __shared__ float    red[16][8];
    __shared__ float    al_s[16], rmax[16], rinv[16];

    const int tid     = threadIdx.x;
    const int bt      = blockIdx.y;
    const int t       = bt % T_;
    const int b       = bt / T_;
    const int rowbase = blockIdx.x * 16;
    const size_t nodebase = (size_t)bt * N_;
    const _Float16* hsrc = hbT + (size_t)bt * H_ * N_;   // [hd][node]

    // Kick off async staging of k-tiles 0 and 1 immediately; the copies
    // overlap with the score/softmax passes below. 2 B128 per thread/stage.
    {
#pragma unroll
        for (int kt0 = 0; kt0 < 2; ++kt0)
            for (int i = tid; i < 256; i += 128) {
                int hd = i >> 2, j = i & 3;
                unsigned loff = (unsigned)(uintptr_t)&hstage[kt0][hd][j * 8];
                unsigned long long ga = (unsigned long long)(uintptr_t)
                    (hsrc + (size_t)hd * N_ + kt0 * 32 + j * 8);
                async_copy_b128(loff, ga);
            }
    }

    for (int i = tid; i < N_; i += 128) ar_s[i] = attn_r[nodebase + i];
    for (int i = tid; i < 16 * 32; i += 128)
        adj_s[i >> 5][i & 31] = adj[(rowbase + (i >> 5)) * 32 + (i & 31)];
    if (tid < 16) al_s[tid] = attn_l[nodebase + rowbase + tid];
    __syncthreads();

    // Pass 1: masked leaky-relu scores, 8 columns per thread iteration.
    for (int i = tid; i < 16 * (N_ / 8); i += 128) {
        int r = i >> 7, c0 = (i & 127) * 8;
        float al = al_s[r];
        unsigned w = adj_s[r][c0 >> 5];
        v8h pk;
#pragma unroll
        for (int j = 0; j < 8; ++j) {
            int c = c0 + j;
            float ev = al + ar_s[c];
            ev = (ev >= 0.f) ? ev : 0.2f * ev;
            bool on = (w >> (c & 31)) & 1u;
            pk[j] = on ? (_Float16)ev : (_Float16)(-INFINITY);
        }
        *(v8h*)&ebuf[r][c0] = pk;
    }
    __syncthreads();

    // Row max: 8 threads per row, vectorized.
    {
        int r = tid >> 3, s = tid & 7;
        float m = -INFINITY;
        for (int c8 = 0; c8 < 16; ++c8) {
            v8h v = *(const v8h*)&ebuf[r][s * 128 + c8 * 8];
#pragma unroll
            for (int j = 0; j < 8; ++j) m = fmaxf(m, (float)v[j]);
        }
        red[r][s] = m;
    }
    __syncthreads();
    if (tid < 16) {
        float m = red[tid][0];
#pragma unroll
        for (int s = 1; s < 8; ++s) m = fmaxf(m, red[tid][s]);
        rmax[tid] = m;
    }
    __syncthreads();

    // Pass 2: unnormalized alpha = exp(e - max) and row sums.
    {
        int r = tid >> 3, s = tid & 7;
        float m = rmax[r], sum = 0.f;
        for (int c8 = 0; c8 < 16; ++c8) {
            v8h v = *(const v8h*)&ebuf[r][s * 128 + c8 * 8];
            v8h o;
#pragma unroll
            for (int j = 0; j < 8; ++j) {
                float a = __expf((float)v[j] - m);   // exp(-inf) -> 0
                o[j] = (_Float16)a;
                sum += a;
            }
            *(v8h*)&ebuf[r][s * 128 + c8 * 8] = o;
        }
        red[r][s] = sum;
    }
    __syncthreads();
    if (tid < 16) {
        float s = 0.f;
#pragma unroll
        for (int j = 0; j < 8; ++j) s += red[tid][j];
        rinv[tid] = 1.0f / s;
    }

    // Pass 3: D[16x64] = alpha[16x1024] @ h[1024x64]; wave w owns n-tile w.
    const int wave = tid >> 5;
    const int lane = tid & 31;
    const int half = lane >> 4;
    const int lm   = lane & 15;
    v8f acc = v8f{};

    for (int kt = 0; kt < N_ / 32; ++kt) {
        const int cur = kt & 1;
        // In-order async completion: keep <=1 stage (2 ops) in flight.
        if (kt < 31) asm volatile("s_wait_asynccnt 0x2" ::: "memory");
        else         asm volatile("s_wait_asynccnt 0x0" ::: "memory");
        __syncthreads();   // stage `kt` visible to all waves (and pass2 done)

        v16h a  = cat8(*(const v8h*)&ebuf[lm][kt * 32 + half * 8],
                       *(const v8h*)&ebuf[lm][kt * 32 + 16 + half * 8]);
        v16h bf = cat8(*(const v8h*)&hstage[cur][wave * 16 + lm][half * 8],
                       *(const v8h*)&hstage[cur][wave * 16 + lm][16 + half * 8]);
        acc = __builtin_amdgcn_wmma_f32_16x16x32_f16(
            false, a, false, bf, (short)0, acc, false, false);

        __syncthreads();   // all waves done reading buffer `cur`
        if (kt + 2 < N_ / 32) {
            for (int i = tid; i < 256; i += 128) {
                int hd = i >> 2, j = i & 3;
                unsigned loff = (unsigned)(uintptr_t)&hstage[cur][hd][j * 8];
                unsigned long long ga = (unsigned long long)(uintptr_t)
                    (hsrc + (size_t)hd * N_ + (kt + 2) * 32 + j * 8);
                async_copy_b128(loff, ga);
            }
        }
    }

    // Epilogue: normalize, bias, exact GeLU; out[b, node, t, hdim].
#pragma unroll
    for (int v = 0; v < 8; ++v) {
        int m    = v + 8 * half;
        int hd   = wave * 16 + lm;
        int node = rowbase + m;
        float val = acc[v] * rinv[m] + bias[hd];
        float g   = 0.5f * val * (1.0f + erff(val * 0.70710678118f));
        out[((((size_t)b * N_ + node) * T_) + t) * H_ + hd] = g;
    }
}

// ---------------------------------------------------------------------------
// Host launcher
// ---------------------------------------------------------------------------
extern "C" void kernel_launch(void* const* d_in, const int* in_sizes, int n_in,
                              void* d_out, int out_size, void* d_ws, size_t ws_size,
                              hipStream_t stream) {
    const float*     x    = (const float*)d_in[0];
    const long long* ei   = (const long long*)d_in[1];
    const float*     W    = (const float*)d_in[2];
    const float*     a_l  = (const float*)d_in[3];
    const float*     a_r  = (const float*)d_in[4];
    const float*     bias = (const float*)d_in[5];
    float*           out  = (float*)d_out;

    char* ws = (char*)d_ws;
    _Float16* hbT    = (_Float16*)ws;                          // 6,291,456 B
    float*    attn_l = (float*)(ws + (size_t)M_ * H_ * 2);     //   196,608 B
    float*    attn_r = attn_l + M_;                            //   196,608 B
    unsigned* adj    = (unsigned*)(ws + (size_t)M_ * H_ * 2 + 2 * (size_t)M_ * 4);

    const int E = in_sizes[1] / 2;                             // 33792

    gat_zero_adj<<<(N_ * N_ / 32) / 256, 256, 0, stream>>>(adj);
    gat_edges<<<(E + 255) / 256, 256, 0, stream>>>(ei, adj, E);
    gat_gemm_h<<<M_ / 64, 128, 0, stream>>>(x, W, hbT);
    dim3 agrid(N_ / 256, BT_);
    gat_attn<<<agrid, 256, 0, stream>>>(hbT, a_l, a_r, attn_l, attn_r);
    dim3 grid(N_ / 16, BT_);
    gat_agg<<<grid, 128, 0, stream>>>(hbT, attn_l, attn_r, adj, bias, out);
}
